// POSModel_14826227106203
// MI455X (gfx1250) — compile-verified, matrix operand order checked by reading, not run on hardware
//
#include <hip/hip_runtime.h>
#include <math.h>

#define S_LEN 8192
#define LC 24
#define WORD_DIM 128
#define CHAR_DIM 24
#define CNN_F 5
#define HID 128
#define NTAGS 45
#define DIN_PAD 160     // 133 padded to 5 K-tiles of 32
#define GATES 512       // 4*HID
#define NCOMB 1024      // fwd+bwd gates
#define OUTD 256        // 2*HID
#define NPAD 48         // 45 padded to 3 N-tiles of 16

typedef __attribute__((ext_vector_type(16))) __bf16 v16bf;
typedef __attribute__((ext_vector_type(8)))  float  v8f;
typedef __attribute__((ext_vector_type(4)))  unsigned int u32x4;

union FragBF16 { v16bf v; u32x4 q[2]; };

__device__ __forceinline__ unsigned short f2bf(float f) {
    unsigned int u = __float_as_uint(f);
    u += 0x7FFFu + ((u >> 16) & 1u);   // round-to-nearest-even
    return (unsigned short)(u >> 16);
}
__device__ __forceinline__ float sigf(float x) {
    return 1.0f / (1.0f + __expf(-x));
}

// ---------------------------------------------------------------------------
// Kernel 1: word-embedding gather + char conv (K=3, pad 1) + length-masked
// max-pool -> rep row as bf16, zero-padded to DIN_PAD.
// One block (128 threads) per word.
// ---------------------------------------------------------------------------
__global__ void build_rep_kernel(const int* __restrict__ word_ids,
                                 const int* __restrict__ char_ids,
                                 const int* __restrict__ char_len,
                                 const float* __restrict__ word_emb,
                                 const float* __restrict__ char_emb,
                                 const float* __restrict__ conv_w,
                                 const float* __restrict__ conv_b,
                                 unsigned short* __restrict__ repA) {
    const int s = blockIdx.x;
    const int tid = threadIdx.x;
    __shared__ float ce[LC][CHAR_DIM];       // [pos][chan]
    __shared__ float cw[CNN_F * CHAR_DIM * 3];
    __shared__ float pool[CNN_F][LC];

    for (int i = tid; i < CNN_F * CHAR_DIM * 3; i += blockDim.x) cw[i] = conv_w[i];
    if (tid < LC) {
        const int cid = char_ids[s * LC + tid];   // pad positions -> zero row
        #pragma unroll
        for (int c = 0; c < CHAR_DIM; ++c) ce[tid][c] = char_emb[cid * CHAR_DIM + c];
    }
    __syncthreads();

    const int len = char_len[s];
    if (tid < LC) {
        float acc[CNN_F] = {0.f, 0.f, 0.f, 0.f, 0.f};
        #pragma unroll
        for (int k = 0; k < 3; ++k) {
            const int tt = tid + k - 1;
            if (tt >= 0 && tt < LC) {
                #pragma unroll
                for (int c = 0; c < CHAR_DIM; ++c) {
                    const float x = ce[tt][c];
                    #pragma unroll
                    for (int f = 0; f < CNN_F; ++f)
                        acc[f] = fmaf(x, cw[(f * CHAR_DIM + c) * 3 + k], acc[f]);
                }
            }
        }
        #pragma unroll
        for (int f = 0; f < CNN_F; ++f)
            pool[f][tid] = (tid < len) ? acc[f] : -INFINITY;
    }
    __syncthreads();

    const int wid = word_ids[s];
    repA[s * DIN_PAD + tid] = f2bf(word_emb[wid * WORD_DIM + tid]);   // tid<128
    if (tid < CNN_F) {
        float m = -INFINITY;
        #pragma unroll
        for (int t = 0; t < LC; ++t) m = fmaxf(m, pool[tid][t]);
        repA[s * DIN_PAD + WORD_DIM + tid] = f2bf(m + conv_b[tid]);
    }
    if (tid < DIN_PAD - (WORD_DIM + CNN_F))                           // 133..159 -> 0
        repA[s * DIN_PAD + WORD_DIM + CNN_F + tid] = 0;
}

// ---------------------------------------------------------------------------
// Weight packing kernels (bf16 conversion, zero padding, bias fusion)
// ---------------------------------------------------------------------------
__global__ void pack_wih_kernel(const float* __restrict__ wf,
                                const float* __restrict__ wb,
                                unsigned short* __restrict__ Wc) {
    const int idx = blockIdx.x * blockDim.x + threadIdx.x;   // NCOMB*DIN_PAD
    if (idx >= NCOMB * DIN_PAD) return;
    const int row = idx / DIN_PAD, k = idx % DIN_PAD;
    float v = 0.0f;
    if (k < 133) v = (row < GATES) ? wf[row * 133 + k] : wb[(row - GATES) * 133 + k];
    Wc[idx] = f2bf(v);
}

__global__ void pack_bias_kernel(const float* __restrict__ bihf, const float* __restrict__ bhhf,
                                 const float* __restrict__ bihb, const float* __restrict__ bhhb,
                                 float* __restrict__ biasc) {
    const int i = blockIdx.x * blockDim.x + threadIdx.x;
    if (i >= NCOMB) return;
    biasc[i] = (i < GATES) ? (bihf[i] + bhhf[i]) : (bihb[i - GATES] + bhhb[i - GATES]);
}

__global__ void pack_whh_kernel(const float* __restrict__ whf,
                                const float* __restrict__ whb,
                                unsigned int* __restrict__ whhp) {   // [2][512][64] bf16x2
    const int idx = blockIdx.x * blockDim.x + threadIdx.x;
    if (idx >= 2 * GATES * (HID / 2)) return;
    const int dir = idx / (GATES * (HID / 2));
    const int rem = idx % (GATES * (HID / 2));
    const int row = rem / (HID / 2), i = rem % (HID / 2);
    const float* w = dir ? whb : whf;
    const unsigned int lo = f2bf(w[row * HID + 2 * i]);
    const unsigned int hi = f2bf(w[row * HID + 2 * i + 1]);
    whhp[idx] = (hi << 16) | lo;
}

__global__ void pack_outw_kernel(const float* __restrict__ ow,
                                 unsigned short* __restrict__ owb) {  // [48][256]
    const int idx = blockIdx.x * blockDim.x + threadIdx.x;
    if (idx >= NPAD * OUTD) return;
    const int row = idx / OUTD, k = idx % OUTD;
    owb[idx] = f2bf(row < NTAGS ? ow[row * OUTD + k] : 0.0f);
}

// ---------------------------------------------------------------------------
// Kernel 2: pre = rep @ Wcᵀ + bias  via v_wmma_f32_16x16x32_bf16.
// One wave per 16x16 tile; K = 160 (5 tiles).
// A layout per ISA: lane<16 holds K=[0..7]∪[16..23]; lane>=16 K=[8..15]∪[24..31].
// B stored [N][K] so it loads identically.
// ---------------------------------------------------------------------------
__global__ void pre_gemm_wmma_kernel(const unsigned short* __restrict__ repA,   // [S][160]
                                     const unsigned short* __restrict__ Wc,     // [1024][160]
                                     const float* __restrict__ biasc,
                                     float* __restrict__ pre) {                 // [S][1024]
    const int tileM = blockIdx.x;
    const int tileN = blockIdx.y;
    const int lane = threadIdx.x;
    const int lm = lane & 15;
    const int kb = (lane >> 4) * 8;
    const unsigned short* arow = repA + (size_t)(tileM * 16 + lm) * DIN_PAD + kb;
    const unsigned short* brow = Wc   + (size_t)(tileN * 16 + lm) * DIN_PAD + kb;

    v8f acc = {0.f, 0.f, 0.f, 0.f, 0.f, 0.f, 0.f, 0.f};
    #pragma unroll
    for (int kt = 0; kt < DIN_PAD / 32; ++kt) {
        FragBF16 a, b;
        a.q[0] = *(const u32x4*)(arow + kt * 32);
        a.q[1] = *(const u32x4*)(arow + kt * 32 + 16);
        b.q[0] = *(const u32x4*)(brow + kt * 32);
        b.q[1] = *(const u32x4*)(brow + kt * 32 + 16);
        acc = __builtin_amdgcn_wmma_f32_16x16x32_bf16(false, a.v, false, b.v,
                                                      (short)0, acc, false, false);
    }
    const int n = tileN * 16 + lm;
    const float bia = biasc[n];
    const int mbase = tileM * 16 + ((lane >> 4) ? 8 : 0);
    #pragma unroll
    for (int r = 0; r < 8; ++r)
        pre[(size_t)(mbase + r) * NCOMB + n] = acc[r] + bia;
}

// ---------------------------------------------------------------------------
// Kernel 3: sequential LSTM, one workgroup (512 threads) per direction.
// Each thread owns one gate row of w_hh, register-resident as 64 bf16x2 words.
// h broadcast from LDS (conflict-free same-address reads).
// ---------------------------------------------------------------------------
__global__ void lstm_seq_kernel(const float* __restrict__ pre,        // [S][1024]
                                const unsigned int* __restrict__ whhp,// [2][512][64]
                                unsigned short* __restrict__ hcat) {  // [S][256] bf16
    const int dir = blockIdx.x;
    const int g = threadIdx.x;                // 0..511
    __shared__ float h_lds[HID];
    __shared__ float z_lds[GATES];

    const unsigned int* wrow = whhp + (size_t)(dir * GATES + g) * (HID / 2);
    unsigned int wreg[HID / 2];
    #pragma unroll
    for (int i = 0; i < HID / 2; ++i) wreg[i] = wrow[i];

    if (g < HID) h_lds[g] = 0.0f;
    float c = 0.0f;
    __syncthreads();

    for (int step = 0; step < S_LEN; ++step) {
        const int t = (dir == 0) ? step : (S_LEN - 1 - step);
        float acc = pre[(size_t)t * NCOMB + dir * GATES + g];
        #pragma unroll
        for (int i = 0; i < HID / 2; ++i) {
            const unsigned int p = wreg[i];
            const float w0 = __uint_as_float(p << 16);
            const float w1 = __uint_as_float(p & 0xFFFF0000u);
            acc = fmaf(w0, h_lds[2 * i], acc);
            acc = fmaf(w1, h_lds[2 * i + 1], acc);
        }
        z_lds[g] = acc;
        __syncthreads();
        if (g < HID) {
            const float iv = sigf(z_lds[g]);
            const float fv = sigf(z_lds[g + HID]);
            const float gv = tanhf(z_lds[g + 2 * HID]);
            const float ov = sigf(z_lds[g + 3 * HID]);
            c = fv * c + iv * gv;
            const float h = ov * tanhf(c);
            h_lds[g] = h;
            hcat[(size_t)t * OUTD + dir * HID + g] = f2bf(h);
        }
        __syncthreads();
    }
}

// ---------------------------------------------------------------------------
// Kernel 4: logits = hcat @ out_wᵀ + out_b  via WMMA bf16. K=256 (8 tiles).
// ---------------------------------------------------------------------------
__global__ void out_gemm_wmma_kernel(const unsigned short* __restrict__ hcat,  // [S][256]
                                     const unsigned short* __restrict__ owb,   // [48][256]
                                     const float* __restrict__ out_b,
                                     float* __restrict__ logits) {             // [S][48]
    const int tileM = blockIdx.x;
    const int tileN = blockIdx.y;   // 0..2
    const int lane = threadIdx.x;
    const int lm = lane & 15;
    const int kb = (lane >> 4) * 8;
    const unsigned short* arow = hcat + (size_t)(tileM * 16 + lm) * OUTD + kb;
    const unsigned short* brow = owb  + (size_t)(tileN * 16 + lm) * OUTD + kb;

    v8f acc = {0.f, 0.f, 0.f, 0.f, 0.f, 0.f, 0.f, 0.f};
    #pragma unroll
    for (int kt = 0; kt < OUTD / 32; ++kt) {
        FragBF16 a, b;
        a.q[0] = *(const u32x4*)(arow + kt * 32);
        a.q[1] = *(const u32x4*)(arow + kt * 32 + 16);
        b.q[0] = *(const u32x4*)(brow + kt * 32);
        b.q[1] = *(const u32x4*)(brow + kt * 32 + 16);
        acc = __builtin_amdgcn_wmma_f32_16x16x32_bf16(false, a.v, false, b.v,
                                                      (short)0, acc, false, false);
    }
    const int n = tileN * 16 + lm;
    const float bia = (n < NTAGS) ? out_b[n] : 0.0f;
    const int mbase = tileM * 16 + ((lane >> 4) ? 8 : 0);
    #pragma unroll
    for (int r = 0; r < 8; ++r)
        logits[(size_t)(mbase + r) * NPAD + n] = acc[r] + bia;
}

// ---------------------------------------------------------------------------
// Kernel 5: row softmax over 45 tags, one wave per row.
// ---------------------------------------------------------------------------
__global__ void softmax_kernel(const float* __restrict__ logits,   // [S][48]
                               float* __restrict__ out) {          // [S][45]
    const int s = blockIdx.x;
    const int lane = threadIdx.x;   // 32
    const float v0 = (lane < NTAGS) ? logits[(size_t)s * NPAD + lane] : -INFINITY;
    const float v1 = (lane + 32 < NTAGS) ? logits[(size_t)s * NPAD + lane + 32] : -INFINITY;
    float m = fmaxf(v0, v1);
    #pragma unroll
    for (int off = 16; off > 0; off >>= 1) m = fmaxf(m, __shfl_xor(m, off, 32));
    const float e0 = (lane < NTAGS) ? __expf(v0 - m) : 0.0f;
    const float e1 = (lane + 32 < NTAGS) ? __expf(v1 - m) : 0.0f;
    float ssum = e0 + e1;
    #pragma unroll
    for (int off = 16; off > 0; off >>= 1) ssum += __shfl_xor(ssum, off, 32);
    const float inv = 1.0f / ssum;
    if (lane < NTAGS)      out[(size_t)s * NTAGS + lane] = e0 * inv;
    if (lane + 32 < NTAGS) out[(size_t)s * NTAGS + lane + 32] = e1 * inv;
}

// ---------------------------------------------------------------------------
extern "C" void kernel_launch(void* const* d_in, const int* in_sizes, int n_in,
                              void* d_out, int out_size, void* d_ws, size_t ws_size,
                              hipStream_t stream) {
    const int*   word_ids  = (const int*)  d_in[0];
    const int*   char_ids  = (const int*)  d_in[1];
    const int*   char_len  = (const int*)  d_in[2];
    const float* word_emb  = (const float*)d_in[3];
    const float* char_emb  = (const float*)d_in[4];
    const float* conv_w    = (const float*)d_in[5];
    const float* conv_b    = (const float*)d_in[6];
    const float* w_ih_f    = (const float*)d_in[7];
    const float* w_hh_f    = (const float*)d_in[8];
    const float* b_ih_f    = (const float*)d_in[9];
    const float* b_hh_f    = (const float*)d_in[10];
    const float* w_ih_b    = (const float*)d_in[11];
    const float* w_hh_b    = (const float*)d_in[12];
    const float* b_ih_b    = (const float*)d_in[13];
    const float* b_hh_b    = (const float*)d_in[14];
    const float* out_w     = (const float*)d_in[15];
    const float* out_b     = (const float*)d_in[16];
    float* out = (float*)d_out;

    char* ws = (char*)d_ws;
    unsigned short* repA  = (unsigned short*)(ws + 0);                 // S*160 bf16
    unsigned short* Wc    = (unsigned short*)(ws + 2621440);           // 1024*160 bf16
    float*          biasc = (float*)         (ws + 2949120);           // 1024 f32
    unsigned int*   whhp  = (unsigned int*)  (ws + 2953216);           // 2*512*64 u32
    unsigned short* owb   = (unsigned short*)(ws + 3215360);           // 48*256 bf16
    unsigned short* hcat  = (unsigned short*)(ws + 3239936);           // S*256 bf16
    float*          lgts  = (float*)         (ws + 7434240);           // S*48 f32
    float*          pre   = (float*)         (ws + 9007104);           // S*1024 f32

    build_rep_kernel<<<S_LEN, 128, 0, stream>>>(word_ids, char_ids, char_len,
                                                word_emb, char_emb, conv_w, conv_b, repA);
    pack_wih_kernel <<<(NCOMB * DIN_PAD + 255) / 256, 256, 0, stream>>>(w_ih_f, w_ih_b, Wc);
    pack_bias_kernel<<<(NCOMB + 255) / 256, 256, 0, stream>>>(b_ih_f, b_hh_f, b_ih_b, b_hh_b, biasc);
    pack_whh_kernel <<<(2 * GATES * (HID / 2) + 255) / 256, 256, 0, stream>>>(w_hh_f, w_hh_b, whhp);
    pack_outw_kernel<<<(NPAD * OUTD + 255) / 256, 256, 0, stream>>>(out_w, owb);

    pre_gemm_wmma_kernel<<<dim3(S_LEN / 16, NCOMB / 16), 32, 0, stream>>>(repA, Wc, biasc, pre);
    lstm_seq_kernel<<<2, GATES, 0, stream>>>(pre, whhp, hcat);
    out_gemm_wmma_kernel<<<dim3(S_LEN / 16, NPAD / 16), 32, 0, stream>>>(hcat, owb, out_b, lgts);
    softmax_kernel<<<S_LEN, 32, 0, stream>>>(lgts, out);
}
